// HardNegativeContrastiveLoss_35948876268191
// MI455X (gfx1250) — matrix operand order repaired and checked by reading, not compiled, not run on hardware
//
#include <hip/hip_runtime.h>
#include <hip/hip_bf16.h>
#include <stdint.h>

#define N_TOTAL 8192
#define HDIM 256
#define NUM_CLASSES 13
#define MARGIN 0.5f
#define IGNORE_INDEX (-100)
#define BIGF 1e9f

#define MTILE 128                        // rows per block (8 waves x 16)
#define NTILE 64                         // columns per LDS tile
#define NSPLIT 8                         // column-dimension split (blockIdx.y)
#define COLS_PER_SPLIT (N_TOTAL / NSPLIT)            // 1024
#define TILES_PER_SPLIT (COLS_PER_SPLIT / NTILE)     // 16
#define LDS_STRIDE (HDIM + 8)            // 264 bf16 = 132 dwords -> 4-bank row stride, conflict-free b128
#define TILE_ELEMS (NTILE * LDS_STRIDE)  // bf16 elements per buffer

typedef __bf16 bf16x8  __attribute__((ext_vector_type(8)));
typedef __bf16 bf16x16 __attribute__((ext_vector_type(16)));
typedef float  f32x8   __attribute__((ext_vector_type(8)));
union Frag { bf16x16 v; bf16x8 h[2]; };

__device__ __forceinline__ int clipc(int a) {
    return a < 0 ? 0 : (a > NUM_CLASSES - 1 ? NUM_CLASSES - 1 : a);
}

// Order-preserving float <-> uint map (unsigned compare == float compare)
__device__ __forceinline__ unsigned f2u(float f) {
    unsigned b = __float_as_uint(f);
    return (b & 0x80000000u) ? ~b : (b | 0x80000000u);
}
__device__ __forceinline__ float u2f(unsigned u) {
    return (u & 0x80000000u) ? __uint_as_float(u ^ 0x80000000u) : __uint_as_float(~u);
}

// ---- CDNA5 async global->LDS (ASYNCcnt-tracked) -----------------------------
__device__ __forceinline__ void async_load_b128(uint32_t lds_addr, const void* gaddr) {
    asm volatile("global_load_async_to_lds_b128 %0, %1, off"
                 :: "v"(lds_addr), "v"((uint64_t)(uintptr_t)gaddr) : "memory");
}
__device__ __forceinline__ void async_load_b32(uint32_t lds_addr, const void* gaddr) {
    asm volatile("global_load_async_to_lds_b32 %0, %1, off"
                 :: "v"(lds_addr), "v"((uint64_t)(uintptr_t)gaddr) : "memory");
}
__device__ __forceinline__ void wait_async0() {
    asm volatile("s_wait_asynccnt 0x0" ::: "memory");
}
__device__ __forceinline__ uint32_t lds_off(const void* p) {
    return (uint32_t)(uintptr_t)p;   // flat LDS addr: DS offset in low 32 bits
}

// Init per-row partial arrays + scalar accumulators
__global__ __launch_bounds__(256)
void init_kernel(unsigned* hpU, unsigned* hnU, float* gsum, unsigned* gcnt) {
    const int i = blockIdx.x * 256 + threadIdx.x;
    hpU[i] = f2u(BIGF);
    hnU[i] = f2u(-BIGF);
    if (i == 0) { *gsum = 0.0f; *gcnt = 0u; }
}

// One block (256 threads) per row: f32 L2-normalize, emit bf16
__global__ __launch_bounds__(256)
void normalize_kernel(const float* __restrict__ e, __bf16* __restrict__ out) {
    __shared__ float red[256];
    const int row = blockIdx.x;
    const int t   = threadIdx.x;
    float x = e[(size_t)row * HDIM + t];
    red[t] = x * x;
    __syncthreads();
    for (int s = 128; s > 0; s >>= 1) {
        if (t < s) red[t] += red[t + s];
        __syncthreads();
    }
    float inv = 1.0f / fmaxf(sqrtf(red[0]), 1e-12f);
    out[(size_t)row * HDIM + t] = (__bf16)(x * inv);
}

// Issue all async loads for one column tile (8 x b128 per thread + labels)
__device__ __forceinline__ void issue_tile_prefetch(const __bf16* __restrict__ E,
                                                    const int* __restrict__ labels,
                                                    __bf16* shBbuf, int* shLabBuf,
                                                    int colBase, int tid) {
    #pragma unroll
    for (int it = 0; it < (NTILE * (HDIM / 8)) / 256; ++it) {   // 8 chunks/thread
        const int i = it * 256 + tid;
        const int r = i >> 5;           // 32 x 16B chunks per row
        const int c = i & 31;
        async_load_b128(lds_off(shBbuf + r * LDS_STRIDE + c * 8),
                        E + (size_t)(colBase + r) * HDIM + c * 8);
    }
    if (tid < NTILE)
        async_load_b32(lds_off(shLabBuf + tid), labels + colBase + tid);
}

// Fused sim-GEMM + masked hardest-pos/neg partial reduction over a column split.
__global__ __launch_bounds__(256)
void hnc_main_kernel(const __bf16* __restrict__ E, const int* __restrict__ labels,
                     unsigned* __restrict__ hpU, unsigned* __restrict__ hnU) {
    extern __shared__ __align__(16) char smem[];
    __bf16* shB   = (__bf16*)smem;                                   // 2 x TILE_ELEMS
    int*    shLab = (int*)(smem + 2 * TILE_ELEMS * sizeof(__bf16));  // 2 x NTILE

    const int tid  = threadIdx.x;
    const int lane = tid & 31;
    const int wid  = tid >> 5;
    const int half = lane >> 4;
    const int l16  = lane & 15;

    const int rowBase  = blockIdx.x * MTILE + wid * 16;
    const int colStart = blockIdx.y * COLS_PER_SPLIT;

    // A fragments resident for the whole kernel (K=256 -> 8 k-steps, 64 VGPRs).
    const int m = rowBase + l16;
    Frag a[8];
    #pragma unroll
    for (int ks = 0; ks < 8; ++ks) {
        const int k0 = ks * 32 + half * 8;
        a[ks].h[0] = *(const bf16x8*)(E + (size_t)m * HDIM + k0);
        a[ks].h[1] = *(const bf16x8*)(E + (size_t)m * HDIM + k0 + 16);
    }

    // Loop-invariant per-row data: raw label + confused partner (sentinel -2).
    int labI[8], partnerI[8];
    #pragma unroll
    for (int r = 0; r < 8; ++r) {
        labI[r] = labels[rowBase + half * 8 + r];
        const int lc = clipc(labI[r]);
        partnerI[r] = (lc >= 1 && lc <= 6) ? (((lc - 1) ^ 1) + 1) : -2;
    }

    float hp[8], hn[8];
    #pragma unroll
    for (int r = 0; r < 8; ++r) { hp[r] = BIGF; hn[r] = -BIGF; }

    issue_tile_prefetch(E, labels, shB, shLab, colStart, tid);

    for (int ct = 0; ct < TILES_PER_SPLIT; ++ct) {
        wait_async0();         // my async writes for tile ct are in LDS
        __syncthreads();       // everyone's are; prev readers of other buffer done
        const int cur = ct & 1;
        if (ct + 1 < TILES_PER_SPLIT)
            issue_tile_prefetch(E, labels,
                                shB + (cur ^ 1) * TILE_ELEMS, shLab + (cur ^ 1) * NTILE,
                                colStart + (ct + 1) * NTILE, tid);

        const __bf16* curB    = shB + cur * TILE_ELEMS;
        const int*    curLab  = shLab + cur * NTILE;
        const int     colBase = colStart + ct * NTILE;

        #pragma unroll
        for (int sub = 0; sub < NTILE / 16; ++sub) {
            const int jr = sub * 16 + l16;       // B column = lane%16
            f32x8 c = {};
            #pragma unroll
            for (int ks = 0; ks < 8; ++ks) {
                Frag b;
                const int k0 = ks * 32 + half * 8;
                b.h[0] = *(const bf16x8*)(curB + jr * LDS_STRIDE + k0);
                b.h[1] = *(const bf16x8*)(curB + jr * LDS_STRIDE + k0 + 16);
                c = __builtin_amdgcn_wmma_f32_16x16x32_bf16(
                        false, a[ks].v, false, b.v, (short)0, c, false, false);
            }
            // Branchless fused epilogue.
            const int labJraw = curLab[sub * 16 + l16];
            const int labJ    = (labJraw != IGNORE_INDEX) ? labJraw : -1; // matches nothing
            const int selfD   = (colBase + sub * 16 + l16) - (rowBase + half * 8);
            #pragma unroll
            for (int r = 0; r < 8; ++r) {
                const bool pe = (labJ == labI[r]) & (selfD != r);
                const bool ne = (labJ == partnerI[r]);
                hp[r] = fminf(hp[r], pe ? c[r] : BIGF);
                hn[r] = fmaxf(hn[r], ne ? c[r] : -BIGF);
            }
        }
    }

    // Reduce across the 16 lanes sharing each row (xor 1,2,4,8 stays in-half)
    #pragma unroll
    for (int r = 0; r < 8; ++r) {
        #pragma unroll
        for (int off = 1; off < 16; off <<= 1) {
            hp[r] = fminf(hp[r], __shfl_xor(hp[r], off, 32));
            hn[r] = fmaxf(hn[r], __shfl_xor(hn[r], off, 32));
        }
    }

    if (l16 == 0) {   // lanes 0 (rows 0-7) and 16 (rows 8-15)
        #pragma unroll
        for (int r = 0; r < 8; ++r) {
            const int row = rowBase + half * 8 + r;
            atomicMin(&hpU[row], f2u(hp[r]));   // global_atomic_min_u32
            atomicMax(&hnU[row], f2u(hn[r]));   // global_atomic_max_u32
        }
    }
}

// Per-row triplet contribution + block reduction
__global__ __launch_bounds__(256)
void reduce_rows_kernel(const unsigned* __restrict__ hpU, const unsigned* __restrict__ hnU,
                        const int* __restrict__ labels,
                        float* __restrict__ gsum, unsigned* __restrict__ gcnt) {
    __shared__ float    ssum[256];
    __shared__ unsigned scnt[256];
    const int t = threadIdx.x;
    const int i = blockIdx.x * 256 + t;

    const float hp = u2f(hpU[i]);
    const float hn = u2f(hnU[i]);
    const int   li = labels[i];
    const bool validI = (li != IGNORE_INDEX);
    const int  lic    = clipc(li);
    const bool anchor = validI && (lic >= 1) && (lic <= 6);   // CONF row nonempty
    const bool hasPos = hp <  0.5f * BIGF;
    const bool hasNeg = hn > -0.5f * BIGF;

    float    tv = 0.0f;
    unsigned cv = 0u;
    if (anchor && hasPos && hasNeg) {
        tv = fmaxf(0.0f, MARGIN + hn - hp);
        cv = 1u;
    }
    ssum[t] = tv; scnt[t] = cv;
    __syncthreads();
    for (int s = 128; s > 0; s >>= 1) {
        if (t < s) { ssum[t] += ssum[t + s]; scnt[t] += scnt[t + s]; }
        __syncthreads();
    }
    if (t == 0) {
        atomicAdd(gsum, ssum[0]);
        atomicAdd(gcnt, scnt[0]);
    }
}

__global__ void finalize_kernel(const float* gsum, const unsigned* gcnt, float* out) {
    const unsigned c = *gcnt;
    out[0] = (c > 0u) ? (*gsum / (float)c) : 0.0f;
}

extern "C" void kernel_launch(void* const* d_in, const int* in_sizes, int n_in,
                              void* d_out, int out_size, void* d_ws, size_t ws_size,
                              hipStream_t stream) {
    const float* emb    = (const float*)d_in[0];   // [16,512,256] f32
    const int*   labels = (const int*)d_in[1];     // [16,512] int

    char*     wsb  = (char*)d_ws;
    __bf16*   Ebf  = (__bf16*)wsb;                 // 4 MB normalized bf16
    size_t    off  = (size_t)N_TOTAL * HDIM * sizeof(__bf16);
    unsigned* hpU  = (unsigned*)(wsb + off);                    // 32 KB
    unsigned* hnU  = (unsigned*)(wsb + off + N_TOTAL * 4);      // 32 KB
    float*    gsum = (float*)(wsb + off + 2 * (size_t)N_TOTAL * 4);
    unsigned* gcnt = (unsigned*)(gsum + 1);

    const size_t shbytes = 2 * TILE_ELEMS * sizeof(__bf16) + 2 * NTILE * sizeof(int);

    init_kernel<<<N_TOTAL / 256, 256, 0, stream>>>(hpU, hnU, gsum, gcnt);
    normalize_kernel<<<N_TOTAL, 256, 0, stream>>>(emb, Ebf);
    dim3 grid(N_TOTAL / MTILE, NSPLIT, 1);
    hnc_main_kernel<<<grid, 256, shbytes, stream>>>(Ebf, labels, hpU, hnU);
    reduce_rows_kernel<<<N_TOTAL / 256, 256, 0, stream>>>(hpU, hnU, labels, gsum, gcnt);
    finalize_kernel<<<1, 1, 0, stream>>>(gsum, gcnt, (float*)d_out);
}
